// AC_MSA_57844619542563
// MI455X (gfx1250) — matrix operand order repaired and checked by reading, not compiled, not run on hardware
//
#include <hip/hip_runtime.h>
#include <hip/hip_bf16.h>

// ---------------------------------------------------------------------------
// AC-MSA for MI455X (gfx1250, wave32, WMMA).  HBM-bound problem (~440 MB min
// traffic -> ~19us @ 23.3 TB/s vs ~22.5 GFLOP compute), so: single pass over
// qkv converting to f16 (halves GEMM-side traffic), all attention working set
// in LDS (320KB/WGP), every matmul via v_wmma_f32_16x16x32_f16.  Contiguous
// 16B staging chunks use GLOBAL_LOAD_ASYNC_TO_LDS_B128 via inline asm
// (ASYNCcnt + s_wait_asynccnt), since the clang builtin's pointer params use
// the unspellable __device__/__shared__ language address spaces.
// ---------------------------------------------------------------------------

typedef __attribute__((ext_vector_type(16))) _Float16 v16h;
typedef __attribute__((ext_vector_type(8)))  _Float16 v8h;
typedef __attribute__((ext_vector_type(8)))  float    v8f;

#define BATCH 2
#define NTOK  65536
#define CH    192
#define NHEAD 4
#define HD    48          // head dim
#define HDP   64          // head dim padded to multiple of 32 for WMMA K
#define GS    128         // group size (CATEGORY_SIZE)
#define NGRP  (NTOK / GS) // 512
#define NCLU  64          // NUM_TOKENS clusters
#define C3    (3 * CH)    // 576
#define HBLK  256         // tokens per histogram block
#define NBLK  (NTOK / HBLK)

#if defined(__HIP_DEVICE_COMPILE__) && defined(__gfx1250__)
#define HAVE_ASYNC_LDS 1
#else
#define HAVE_ASYNC_LDS 0
#endif

#if HAVE_ASYNC_LDS
// GVS mode: global_mem_addr = SADDR[63:0] + VADDR[31:0]; LDS dest = low 32
// bits of the generic shared pointer.
__device__ inline void async_b128(const void* gbase, unsigned voff_bytes,
                                  void* lds) {
  unsigned ldsoff = (unsigned)(size_t)lds;
  asm volatile("global_load_async_to_lds_b128 %0, %1, %2"
               :: "v"(ldsoff), "v"(voff_bytes), "s"(gbase)
               : "memory");
}
__device__ inline void async_wait0() {
  asm volatile("s_wait_asynccnt 0" ::: "memory");
}
#endif

union F8 { v8f v; float f[8]; };
union H16 { v16h v; v8h h2[2]; _Float16 h[16]; };

// A fragment (16x32 f16, row major in LDS, contiguous along K).
// ISA 7.12.2: lane<16 -> M=lane, halves0..7 = K(k0+0..7), halves8..15 = K(k0+16..23);
//             lane>=16 -> same M set, K offset +8.
__device__ inline v16h load_a_frag(const _Float16* p, int row0, int stride,
                                   int k0, int lane) {
  int m = lane & 15, ks = (lane >> 4) & 1;
  const _Float16* r = p + (size_t)(row0 + m) * stride + k0 + ks * 8;
  H16 u;
  u.h2[0] = *(const v8h*)(r);
  u.h2[1] = *(const v8h*)(r + 16);
  return u.v;
}

// B fragment (32x16 f16).  Stored as [n][k] row major in LDS (contiguous K).
// lane<16 -> N=lane, halves0..7 = K(k0+0..7), halves8..15 = K(k0+16..23); lane>=16 -> K offset +8.
__device__ inline v16h load_b_frag(const _Float16* p, int n0, int stride,
                                   int k0, int lane) {
  int n = lane & 15, ks = (lane >> 4) & 1;
  const _Float16* r = p + (size_t)(n0 + n) * stride + k0 + ks * 8;
  H16 u;
  u.h2[0] = *(const v8h*)(r);
  u.h2[1] = *(const v8h*)(r + 16);
  return u.v;
}

// ---------------- clustering (stable counting sort) ------------------------

__global__ void acmsa_argmax_kernel(const float* __restrict__ sim,
                                    int* __restrict__ tk) {
  int t = blockIdx.x * blockDim.x + threadIdx.x;
  if (t >= BATCH * NTOK) return;
  const float* s = sim + (size_t)t * NCLU;
  float bv = s[0];
  int bi = 0;
  for (int j = 1; j < NCLU; ++j) {
    float v = s[j];
    if (v > bv) { bv = v; bi = j; }   // first max, like jnp.argmax
  }
  tk[t] = bi;
}

__global__ void acmsa_hist_kernel(const int* __restrict__ tk,
                                  int* __restrict__ hist) {
  __shared__ int h[NCLU];
  int t = threadIdx.x;
  int blk = blockIdx.x % NBLK;
  int b = blockIdx.x / NBLK;
  if (t < NCLU) h[t] = 0;
  __syncthreads();
  int c = tk[(size_t)b * NTOK + blk * HBLK + t];
  atomicAdd(&h[c], 1);
  __syncthreads();
  if (t < NCLU) hist[((size_t)(b * NBLK + blk)) * NCLU + t] = h[t];
}

__global__ void acmsa_scan_kernel(int* __restrict__ hist,
                                  int* __restrict__ cbase) {
  int b = blockIdx.x;
  int c = threadIdx.x;   // 64 threads: one per cluster
  int run = 0;
  for (int blk = 0; blk < NBLK; ++blk) {
    size_t idx = ((size_t)(b * NBLK + blk)) * NCLU + c;
    int v = hist[idx];
    hist[idx] = run;     // exclusive per-block prefix within cluster
    run += v;
  }
  __shared__ int tot[NCLU];
  tot[c] = run;
  __syncthreads();
  if (c == 0) {
    int acc = 0;
    for (int i = 0; i < NCLU; ++i) { int v = tot[i]; cbase[b * NCLU + i] = acc; acc += v; }
  }
}

__global__ void acmsa_rank_kernel(const int* __restrict__ tk,
                                  const int* __restrict__ hist,
                                  const int* __restrict__ cbase,
                                  int* __restrict__ rev) {
  __shared__ int cid[HBLK];
  int t = threadIdx.x;
  int blk = blockIdx.x % NBLK;
  int b = blockIdx.x / NBLK;
  int i = blk * HBLK + t;
  int c = tk[(size_t)b * NTOK + i];
  cid[t] = c;
  __syncthreads();
  int lr = 0;
  for (int j = 0; j < t; ++j) lr += (cid[j] == c);   // stable in-block rank
  rev[(size_t)b * NTOK + i] =
      cbase[b * NCLU + c] + hist[((size_t)(b * NBLK + blk)) * NCLU + c] + lr;
}

// scatter qkv into sorted order, fp32 -> f16
__global__ void acmsa_shuffle_kernel(const float* __restrict__ qkv,
                                     const int* __restrict__ rev,
                                     _Float16* __restrict__ shuf) {
  size_t total = (size_t)BATCH * NTOK * C3;
  size_t stride = (size_t)gridDim.x * blockDim.x;
  for (size_t e = (size_t)blockIdx.x * blockDim.x + threadIdx.x; e < total; e += stride) {
    __builtin_prefetch(qkv + e + 4096, 0, 0);   // global_prefetch_b8
    int col = (int)(e % C3);
    size_t tf = e / C3;
    int i = (int)(tf % NTOK);
    int b = (int)(tf / NTOK);
    int r = rev[(size_t)b * NTOK + i];
    shuf[((size_t)b * NTOK + r) * C3 + col] = (_Float16)qkv[e];
  }
}

__global__ void acmsa_wconv_kernel(const float* __restrict__ pw,
                                   _Float16* __restrict__ wf) {
  int e = blockIdx.x * blockDim.x + threadIdx.x;
  if (e < CH * CH) wf[e] = (_Float16)pw[e];   // keep (out,in) layout: B[n=o][k]
}

// ---------------- grouped attention (one WG per b,group,head) --------------

__global__ void acmsa_attn_kernel(const _Float16* __restrict__ shuf,
                                  const float* __restrict__ logit_scale,
                                  _Float16* __restrict__ aout) {
  extern __shared__ __align__(16) char smem[];
  _Float16* Qs = (_Float16*)smem;            // 128 x 64  (hd padded, zeros)
  _Float16* Ks = Qs + GS * HDP;              // 128 x 64
  _Float16* Vt = Ks + GS * HDP;              // 48 x 128  (transposed: [d][tok])
  float*    Sb = (float*)(smem + (2 * GS * HDP + HD * GS) * sizeof(_Float16)); // 128x128 f32
  _Float16* Pp = (_Float16*)Sb;              // reuse: P row r at half-index r*256

  int bid = blockIdx.x;
  int h = bid % NHEAD;
  int g = (bid / NHEAD) % NGRP;
  int b = bid / (NHEAD * NGRP);
  int t = threadIdx.x, lane = t & 31, wave = t >> 5;
  int m0 = wave * 16;

#if HAVE_ASYNC_LDS
  // Q,K payload: per row 48 halves = 6 x 16B chunks, async global->LDS.
  for (int s = t; s < GS * 6; s += blockDim.x) {
    int row = s / 6, c = (s % 6) * 8;   // half offset within row
    unsigned rb = (unsigned)(((unsigned)(b * NTOK + g * GS + row)) * C3 + h * HD + c);
    async_b128(shuf, rb * 2u, Qs + row * HDP + c);
    async_b128(shuf, (rb + CH) * 2u, Ks + row * HDP + c);
  }
  // zero the K-dim padding (cols 48..63) with plain stores
  for (int s = t; s < GS * 16; s += blockDim.x) {
    int row = s >> 4, d = HD + (s & 15);
    Qs[row * HDP + d] = (_Float16)0;
    Ks[row * HDP + d] = (_Float16)0;
  }
#else
  for (int s = t; s < GS * HDP; s += blockDim.x) {
    int row = s >> 6, d = s & 63;
    size_t rb = ((size_t)b * NTOK + (size_t)g * GS + row) * C3 + h * HD;
    _Float16 q = (_Float16)0, k = (_Float16)0;
    if (d < HD) { q = shuf[rb + d]; k = shuf[rb + CH + d]; }
    Qs[s] = q;
    Ks[s] = k;
  }
#endif
  // V transposed ([d][token]) cannot be a contiguous async copy: plain path
  for (int s = t; s < HD * GS; s += blockDim.x) {
    int d = s >> 7, row = s & 127;
    size_t rb = ((size_t)b * NTOK + (size_t)g * GS + row) * C3 + 2 * CH + h * HD;
    Vt[s] = shuf[rb + d];   // s == d*128 + row
  }
#if HAVE_ASYNC_LDS
  async_wait0();
#endif
  __syncthreads();

  // S = Q * K^T   (each wave: 16x128 stripe; 8 tiles x 2 K-steps)
  for (int n0 = 0; n0 < GS; n0 += 16) {
    F8 acc;
    for (int i = 0; i < 8; ++i) acc.f[i] = 0.f;
    for (int k0 = 0; k0 < HDP; k0 += 32) {
      v16h a = load_a_frag(Qs, m0, HDP, k0, lane);
      v16h bb = load_b_frag(Ks, n0, HDP, k0, lane);
      acc.v = __builtin_amdgcn_wmma_f32_16x16x32_f16(false, a, false, bb,
                                                     (short)0, acc.v, false, false);
    }
    int hi = (lane >> 4) & 1, nn = lane & 15;
    for (int i = 0; i < 8; ++i)
      Sb[(size_t)(m0 + i + 8 * hi) * GS + n0 + nn] = acc.f[i];
  }
  __syncthreads();

  // softmax over each wave's 16 rows; write f16 P in place (row r -> Pp[r*256+c])
  float ls = logit_scale[0];
  float scale = __expf(fminf(ls, 4.60517018598809136804f));  // log(1/0.01)
  for (int r = m0; r < m0 + 16; ++r) {
    float x[4], mx = -3.402823466e+38f;
    for (int j = 0; j < 4; ++j) {
      x[j] = Sb[(size_t)r * GS + lane + 32 * j] * scale;
      mx = fmaxf(mx, x[j]);
    }
    for (int d = 16; d >= 1; d >>= 1) mx = fmaxf(mx, __shfl_xor(mx, d, 32));
    float sum = 0.f;
    for (int j = 0; j < 4; ++j) { x[j] = __expf(x[j] - mx); sum += x[j]; }
    for (int d = 16; d >= 1; d >>= 1) sum += __shfl_xor(sum, d, 32);
    float inv = 1.f / sum;
    for (int j = 0; j < 4; ++j)
      Pp[(size_t)r * 256 + lane + 32 * j] = (_Float16)(x[j] * inv);
  }
  __syncthreads();

  // O = P * V    (each wave: 16x48; 3 tiles x 4 K-steps)
  for (int n0 = 0; n0 < HD; n0 += 16) {
    F8 acc;
    for (int i = 0; i < 8; ++i) acc.f[i] = 0.f;
    for (int k0 = 0; k0 < GS; k0 += 32) {
      v16h a = load_a_frag(Pp, m0, 256, k0, lane);
      v16h bb = load_b_frag(Vt, n0, GS, k0, lane);
      acc.v = __builtin_amdgcn_wmma_f32_16x16x32_f16(false, a, false, bb,
                                                     (short)0, acc.v, false, false);
    }
    int hi = (lane >> 4) & 1, nn = lane & 15;
    for (int i = 0; i < 8; ++i) {
      size_t row = (size_t)b * NTOK + (size_t)g * GS + m0 + i + 8 * hi;
      aout[row * CH + h * HD + n0 + nn] = (_Float16)acc.f[i];
    }
  }
}

// ---------------- unshuffle + output projection ----------------------------

__global__ void acmsa_proj_kernel(const _Float16* __restrict__ aout,
                                  const int* __restrict__ rev,
                                  const _Float16* __restrict__ wf,
                                  const float* __restrict__ pb,
                                  float* __restrict__ out) {
  extern __shared__ __align__(16) char smem[];
  _Float16* Xs = (_Float16*)smem;            // 128 x 192 gathered rows
  _Float16* Ws = Xs + GS * CH;               // 192 x 192 (out,in) == B[n][k]
  int* revs = (int*)(smem + (GS * CH + CH * CH) * sizeof(_Float16));

  int nb = NTOK / GS;
  int i0 = (blockIdx.x % nb) * GS;
  int b = blockIdx.x / nb;
  int t = threadIdx.x, lane = t & 31, wave = t >> 5;
  int m0 = wave * 16;

  if (t < GS) revs[t] = rev[(size_t)b * NTOK + i0 + t];
  __syncthreads();
#if HAVE_ASYNC_LDS
  // weights: one contiguous 73728B copy in 16B chunks
  for (int s = t; s < (CH * CH) / 8; s += blockDim.x)
    async_b128(wf, (unsigned)(s * 16), Ws + s * 8);
  // gathered activation rows: 24 x 16B chunks per row (batch base kept uniform)
  const _Float16* abase = aout + (size_t)b * NTOK * CH;
  for (int s = t; s < GS * 24; s += blockDim.x) {
    int r = s / 24, c = (s % 24) * 8;
    unsigned voff = (unsigned)(revs[r] * CH + c) * 2u;
    async_b128(abase, voff, Xs + r * CH + c);
  }
  async_wait0();
#else
  for (int s = t; s < CH * CH; s += blockDim.x) Ws[s] = wf[s];
  for (int s = t; s < GS * CH; s += blockDim.x) {
    int r = s / CH, col = s % CH;
    Xs[s] = aout[((size_t)b * NTOK + revs[r]) * CH + col];
  }
#endif
  __syncthreads();

  for (int n0 = 0; n0 < CH; n0 += 16) {   // 12 tiles x 6 K-steps per wave
    F8 acc;
    for (int i = 0; i < 8; ++i) acc.f[i] = 0.f;
    for (int k0 = 0; k0 < CH; k0 += 32) {
      v16h a = load_a_frag(Xs, m0, CH, k0, lane);
      v16h bb = load_b_frag(Ws, n0, CH, k0, lane);
      acc.v = __builtin_amdgcn_wmma_f32_16x16x32_f16(false, a, false, bb,
                                                     (short)0, acc.v, false, false);
    }
    int hi = (lane >> 4) & 1, nn = lane & 15;
    float bias = pb[n0 + nn];
    for (int i = 0; i < 8; ++i) {
      size_t row = (size_t)b * NTOK + i0 + m0 + i + 8 * hi;
      out[row * CH + n0 + nn] = acc.f[i] + bias;
    }
  }
}

// ---------------------------------------------------------------------------

extern "C" void kernel_launch(void* const* d_in, const int* in_sizes, int n_in,
                              void* d_out, int out_size, void* d_ws, size_t ws_size,
                              hipStream_t stream) {
  (void)in_sizes; (void)n_in; (void)out_size; (void)ws_size;
  const float* qkv = (const float*)d_in[0];
  const float* sim = (const float*)d_in[1];
  const float* pw  = (const float*)d_in[2];
  const float* pb  = (const float*)d_in[3];
  const float* ls  = (const float*)d_in[4];
  float* out = (float*)d_out;

  auto al = [](size_t x) { return (x + 255) & ~(size_t)255; };
  char* ws = (char*)d_ws;
  size_t off = 0;
  int* tk = (int*)(ws + off);        off = al(off + (size_t)BATCH * NTOK * 4);
  int* rev = (int*)(ws + off);       off = al(off + (size_t)BATCH * NTOK * 4);
  int* hist = (int*)(ws + off);      off = al(off + (size_t)BATCH * NBLK * NCLU * 4);
  int* cbase = (int*)(ws + off);     off = al(off + (size_t)BATCH * NCLU * 4);
  _Float16* shuf = (_Float16*)(ws + off); off = al(off + (size_t)BATCH * NTOK * C3 * 2);
  _Float16* aout = (_Float16*)(ws + off); off = al(off + (size_t)BATCH * NTOK * CH * 2);
  _Float16* wf = (_Float16*)(ws + off);   off = al(off + (size_t)CH * CH * 2);

  size_t attn_lds = (size_t)(2 * GS * HDP + HD * GS) * 2 + (size_t)GS * GS * 4;   // 110592
  size_t proj_lds = (size_t)(GS * CH + CH * CH) * 2 + GS * 4;                      // 123392
  (void)hipFuncSetAttribute((const void*)acmsa_attn_kernel,
                            hipFuncAttributeMaxDynamicSharedMemorySize, (int)attn_lds);
  (void)hipFuncSetAttribute((const void*)acmsa_proj_kernel,
                            hipFuncAttributeMaxDynamicSharedMemorySize, (int)proj_lds);

  acmsa_argmax_kernel<<<(BATCH * NTOK + 255) / 256, 256, 0, stream>>>(sim, tk);
  acmsa_hist_kernel<<<BATCH * NBLK, HBLK, 0, stream>>>(tk, hist);
  acmsa_scan_kernel<<<BATCH, NCLU, 0, stream>>>(hist, cbase);
  acmsa_rank_kernel<<<BATCH * NBLK, HBLK, 0, stream>>>(tk, hist, cbase, rev);
  acmsa_shuffle_kernel<<<8192, 256, 0, stream>>>(qkv, rev, shuf);
  acmsa_wconv_kernel<<<(CH * CH + 255) / 256, 256, 0, stream>>>(pw, wf);
  acmsa_attn_kernel<<<BATCH * NGRP * NHEAD, 256, attn_lds, stream>>>(shuf, ls, aout);
  acmsa_proj_kernel<<<BATCH * (NTOK / GS), 256, proj_lds, stream>>>(aout, rev, wf, pb, out);
}
